// TransformerSelfAttnLayer_88974542504505
// MI455X (gfx1250) — matrix operand order repaired and checked by reading, not compiled, not run on hardware
//
#include <hip/hip_runtime.h>
#include <hip/hip_bf16.h>
#include <hip/hip_fp16.h>
#include <stdint.h>

// ---------------------------------------------------------------------------
// TransformerSelfAttnLayer for MI455X (gfx1250, wave32, WMMA + TDM).
//   W=1024, B=16, E=512, H=8, D=64.  All matmuls via v_wmma_f32_16x16x32_f16.
//   GEMMs: TDM (tensor_load_to_lds) double-buffered LDS tile staging,
//          s_wait_tensorcnt + workgroup barrier handshake (wave 0 issues).
//   Attention: 32x1024 score strip LDS-resident (f32) + f16 P buffer
//          (192KB of the 320KB/WGP LDS); softmax denominator deferred to the
//          f32 O-accumulator epilogue.  The 512MB score tensor never hits HBM.
// Workspace (~86 MB): xln | Wqkv_f16 | Wout_f16 | Qh | Kh | Vt | attn_out
// ---------------------------------------------------------------------------

typedef __attribute__((ext_vector_type(16))) _Float16 v16h;
typedef __attribute__((ext_vector_type(8)))  _Float16 v8h;
typedef __attribute__((ext_vector_type(8)))  float    v8f;
typedef __attribute__((ext_vector_type(4)))  unsigned v4u;
typedef __attribute__((ext_vector_type(8)))  int      v8i;
typedef __attribute__((ext_vector_type(4)))  int      v4i;

constexpr int W_LEN = 1024, BSZ = 16, EMBED = 512, NHEADS = 8, HEAD_DIM = 64;
constexpr int NTOK  = W_LEN * BSZ;      // 16384
constexpr int QKV_N = 3 * EMBED;        // 1536
constexpr int NBH   = BSZ * NHEADS;     // 128

#if defined(__AMDGCN__) && __has_builtin(__builtin_amdgcn_tensor_load_to_lds) && \
    __has_builtin(__builtin_amdgcn_s_wait_tensorcnt)
#define USE_TDM 1
#else
#define USE_TDM 0
#endif

__device__ __forceinline__ v8f wmma_f16(v16h a, v16h b, v8f c) {
  return __builtin_amdgcn_wmma_f32_16x16x32_f16(false, a, false, b, (short)0, c,
                                                false, false);
}

// A fragment (16x32 f16): lane holds row m=lane%16; K chunks [g0,g0+8),[g0+16,g0+24)
__device__ __forceinline__ v16h make_a_frag(const _Float16* row_k, int lane) {
  const int g0 = (lane >> 4) << 3;               // 0 or 8
  v8h lo = *(const v8h*)(row_k + g0);
  v8h hi = *(const v8h*)(row_k + g0 + 16);
  return __builtin_shufflevector(lo, hi, 0, 1, 2, 3, 4, 5, 6, 7, 8, 9, 10, 11,
                                 12, 13, 14, 15);
}
// B fragment (32x16 f16): lane holds B^T row n=lane%16; K chunks [g0,g0+8),[g0+16,g0+24)
__device__ __forceinline__ v16h make_b_frag(const _Float16* bt_row_k, int lane) {
  const int g0 = (lane >> 4) << 3;               // 0 or 8
  v8h lo = *(const v8h*)(bt_row_k + g0);
  v8h hi = *(const v8h*)(bt_row_k + g0 + 16);
  return __builtin_shufflevector(lo, hi, 0, 1, 2, 3, 4, 5, 6, 7, 8, 9, 10, 11,
                                 12, 13, 14, 15);
}

#if USE_TDM
// Issue a 2-D TDM tile load: tile (tile_d1 rows x tile_d0 elems) of a 2-byte
// tensor (tensor_d1 x tensor_d0, row stride stride0 elems) into LDS.
// D# bit layout per CDNA5 ISA 8.3/8.4: group0 {count=1, lds_addr, global_addr,
// type=2}; group1 {data_size=2B, dims, tile dims, stride}; groups 2/3 zero (2D).
__device__ __forceinline__ void tdm_load_2d(unsigned lds_addr, const void* gptr,
                                            unsigned tensor_d0, unsigned tensor_d1,
                                            unsigned tile_d0, unsigned tile_d1,
                                            unsigned stride0) {
  const unsigned long long ga = (unsigned long long)(uintptr_t)gptr;
  v4u g0;
  g0.x = 1u;                                           // count=1, user mode
  g0.y = lds_addr;                                     // bits 63:32
  g0.z = (unsigned)ga;                                 // global_addr[31:0]
  g0.w = (unsigned)((ga >> 32) & 0x1FFFFFFu) | (2u << 30);  // addr[56:32]|type=2
  v8i g1;
  g1[0] = (int)(1u << 16);                             // data_size=1 -> 2 bytes
  g1[1] = (int)((tensor_d0 & 0xFFFFu) << 16);          // dim0[15:0] @ bits 63:48
  g1[2] = (int)(((tensor_d0 >> 16) & 0xFFFFu) | ((tensor_d1 & 0xFFFFu) << 16));
  g1[3] = (int)(((tensor_d1 >> 16) & 0xFFFFu) | ((tile_d0 & 0xFFFFu) << 16));
  g1[4] = (int)(tile_d1 & 0xFFFFu);                    // tile_dim1, tile_dim2=0
  g1[5] = (int)stride0;                                // dim0_stride[31:0]
  g1[6] = 0;
  g1[7] = 0;
  const v4i z4 = {0, 0, 0, 0};
#if __clang_major__ >= 23
  const v8i z8 = {0, 0, 0, 0, 0, 0, 0, 0};
  __builtin_amdgcn_tensor_load_to_lds(g0, g1, z4, z4, z8, 0);
#else
  __builtin_amdgcn_tensor_load_to_lds(g0, g1, z4, z4, 0);
#endif
}
#endif

// Shared GEMM main loop: C(64x128) wg tile, K=512, 8 waves (2x4) of 32x32.
// A[ M,512 ] row-major f16, B^T[ N,512 ] row-major f16 (both K-contiguous).
__device__ __forceinline__ void gemm_mainloop_64x128(
    const _Float16* __restrict__ Ag, int a_rows, const _Float16* __restrict__ Bg,
    int b_rows, int m0blk, int n0blk, _Float16* sA, _Float16* sB, int lane,
    int wave, v8f acc[2][2]) {
  const int am = lane & 15;
  const int mw = (wave >> 2) * 32;
  const int nw = (wave & 3) * 32;
#if USE_TDM
  const unsigned ldsA = (unsigned)(uintptr_t)sA;   // generic->LDS offset (addr[31:0])
  const unsigned ldsB = (unsigned)(uintptr_t)sB;
  if (wave == 0) {  // tensor ops ignore EXEC; one issue per wave
    tdm_load_2d(ldsA, Ag + (size_t)m0blk * EMBED, EMBED, a_rows, 32, 64, EMBED);
    tdm_load_2d(ldsB, Bg + (size_t)n0blk * EMBED, EMBED, b_rows, 32, 128, EMBED);
  }
  for (int kk = 0; kk < EMBED; kk += 32) {
    const int buf = (kk >> 5) & 1;
    if (wave == 0) __builtin_amdgcn_s_wait_tensorcnt(0);
    __syncthreads();                         // tiles visible; prev reads done
    if (wave == 0 && kk + 32 < EMBED) {      // prefetch next k-step
      tdm_load_2d(ldsA + (buf ^ 1) * (64 * 32 * 2),
                  Ag + (size_t)m0blk * EMBED + kk + 32, EMBED, a_rows, 32, 64, EMBED);
      tdm_load_2d(ldsB + (buf ^ 1) * (128 * 32 * 2),
                  Bg + (size_t)n0blk * EMBED + kk + 32, EMBED, b_rows, 32, 128, EMBED);
    }
    const _Float16* As = sA + buf * (64 * 32);
    const _Float16* Bs = sB + buf * (128 * 32);
#else
  for (int kk = 0; kk < EMBED; kk += 32) {
    __syncthreads();                         // prev compute done before refill
    {  // cooperative fill, single buffer
      const int t = threadIdx.x;
      const int ra = t >> 2, ca = (t & 3) * 8;    // A: 64x32, v8h per thread
      *(v8h*)(sA + ra * 32 + ca) =
          *(const v8h*)(Ag + (size_t)(m0blk + ra) * EMBED + kk + ca);
      const int rb = t >> 1, cb = (t & 1) * 16;   // B: 128x32, v16h per thread
      *(v16h*)(sB + rb * 32 + cb) =
          *(const v16h*)(Bg + (size_t)(n0blk + rb) * EMBED + kk + cb);
    }
    __syncthreads();
    const _Float16* As = sA;
    const _Float16* Bs = sB;
#endif
    v16h a[2], b[2];
#pragma unroll
    for (int i = 0; i < 2; ++i) {
      a[i] = make_a_frag(As + (mw + i * 16 + am) * 32, lane);
      b[i] = make_b_frag(Bs + (nw + i * 16 + am) * 32, lane);
    }
#pragma unroll
    for (int i = 0; i < 2; ++i)
#pragma unroll
      for (int j = 0; j < 2; ++j) acc[i][j] = wmma_f16(a[i], b[j], acc[i][j]);
  }
}

// ---------------------------------------------------------------- weights f32->f16
__global__ void cvt_kernel(const float* __restrict__ src,
                           _Float16* __restrict__ dst, int n) {
  int i = blockIdx.x * 256 + threadIdx.x;
  if (i < n) dst[i] = (_Float16)src[i];
}

// ---------------------------------------------------------------- LayerNorm -> f16
__global__ __launch_bounds__(256) void ln_kernel(const float* __restrict__ feat,
                                                 const float* __restrict__ g,
                                                 const float* __restrict__ b,
                                                 _Float16* __restrict__ xln) {
  const int lane = threadIdx.x & 31, wave = threadIdx.x >> 5;
  const int tok = blockIdx.x * 8 + wave;
  const float4* x4 = (const float4*)(feat + (size_t)tok * EMBED);
  float4 v[4];
  float s = 0.f, ss = 0.f;
#pragma unroll
  for (int i = 0; i < 4; ++i) {
    v[i] = x4[lane + 32 * i];
    s  += v[i].x + v[i].y + v[i].z + v[i].w;
    ss += v[i].x * v[i].x + v[i].y * v[i].y + v[i].z * v[i].z + v[i].w * v[i].w;
  }
#pragma unroll
  for (int m = 16; m >= 1; m >>= 1) {
    s  += __shfl_xor(s, m, 32);
    ss += __shfl_xor(ss, m, 32);
  }
  const float mu = s * (1.f / EMBED);
  const float rstd = rsqrtf(ss * (1.f / EMBED) - mu * mu + 1e-5f);
  _Float16* o = xln + (size_t)tok * EMBED;
#pragma unroll
  for (int i = 0; i < 4; ++i) {
    const int e = (lane + 32 * i) * 4;
    const float* vv = (const float*)&v[i];
#pragma unroll
    for (int j = 0; j < 4; ++j)
      o[e + j] = (_Float16)((vv[j] - mu) * rstd * g[e + j] + b[e + j]);
  }
}

// ---------------------------------------------------------------- QKV GEMM (WMMA+TDM)
__global__ __launch_bounds__(256) void qkv_gemm_wmma(
    const _Float16* __restrict__ xln, const _Float16* __restrict__ wqkv,
    const float* __restrict__ bias, _Float16* __restrict__ Qh,
    _Float16* __restrict__ Kh, _Float16* __restrict__ Vt) {
  __shared__ alignas(64) _Float16 sA[2 * 64 * 32];
  __shared__ alignas(64) _Float16 sB[2 * 128 * 32];
  const int lane = threadIdx.x & 31, wave = threadIdx.x >> 5;
  v8f acc[2][2] = {};
  gemm_mainloop_64x128(xln, NTOK, wqkv, QKV_N, blockIdx.y * 64, blockIdx.x * 128,
                       sA, sB, lane, wave, acc);
  const int am = lane & 15, mhalf = (lane >> 4) << 3;
  const int m0 = blockIdx.y * 64 + (wave >> 2) * 32;
  const int n0 = blockIdx.x * 128 + (wave & 3) * 32;
#pragma unroll
  for (int i = 0; i < 2; ++i)
#pragma unroll
    for (int j = 0; j < 2; ++j) {
      const int ncol = n0 + j * 16 + am;
      const float bb = bias[ncol];
      const int part = ncol >> 9, oo = ncol & 511;
      const int h = oo >> 6, d = oo & 63;
#pragma unroll
      for (int r = 0; r < 8; ++r) {
        const int t = m0 + i * 16 + r + mhalf;      // token = w*16 + b
        const int wpos = t >> 4, bidx = t & 15;
        const int bh = bidx * NHEADS + h;
        const float val = acc[i][j][r] + bb;
        if (part == 0)
          Qh[((size_t)bh * W_LEN + wpos) * HEAD_DIM + d] = (_Float16)(val * 0.125f);
        else if (part == 1)
          Kh[((size_t)bh * W_LEN + wpos) * HEAD_DIM + d] = (_Float16)val;
        else
          Vt[((size_t)bh * HEAD_DIM + d) * W_LEN + wpos] = (_Float16)val;
      }
    }
}

// ---------------------------------------------------------------- attention
// grid=(bh=128, W/32=32).  S[32][1024]f32 + P[32][1024]f16 LDS-resident.
__global__ __launch_bounds__(256) void attn_kernel(
    const _Float16* __restrict__ Qh, const _Float16* __restrict__ Kh,
    const _Float16* __restrict__ Vt, _Float16* __restrict__ attn_out) {
  extern __shared__ char smem[];
  float* S = (float*)smem;                                        // 128 KB
  _Float16* P = (_Float16*)(smem + 32 * 1024 * sizeof(float));    //  64 KB
  float* inv_sum = (float*)(smem + 32 * 1024 * 4 + 32 * 1024 * 2);
  const int bh = blockIdx.x;
  const int w0 = blockIdx.y * 32;
  const int lane = threadIdx.x & 31, wave = threadIdx.x >> 5;
  const int am = lane & 15, mhalf = (lane >> 4) << 3;
  const _Float16* Qb = Qh + (size_t)bh * W_LEN * HEAD_DIM;
  const _Float16* Kb = Kh + (size_t)bh * W_LEN * HEAD_DIM;
  const _Float16* Vb = Vt + (size_t)bh * HEAD_DIM * W_LEN;

  // ---- Phase 1: scores S = (Q*scale) K^T
  {
    const int rt = wave >> 2, c0 = (wave & 3) * 256;
    v8f acc[16] = {};
#pragma unroll
    for (int kk = 0; kk < HEAD_DIM; kk += 32) {
      v16h a = make_a_frag(Qb + (size_t)(w0 + rt * 16 + am) * HEAD_DIM + kk, lane);
#pragma unroll
      for (int ct = 0; ct < 16; ++ct) {
        v16h b = make_b_frag(Kb + (size_t)(c0 + ct * 16 + am) * HEAD_DIM + kk, lane);
        acc[ct] = wmma_f16(a, b, acc[ct]);
      }
    }
#pragma unroll
    for (int ct = 0; ct < 16; ++ct)
#pragma unroll
      for (int r = 0; r < 8; ++r)
        S[(rt * 16 + r + mhalf) * 1024 + c0 + ct * 16 + am] = acc[ct][r];
  }
  __syncthreads();

  // ---- Phase 2: row softmax -> unnormalized f16 P, 1/sum deferred
  {
    const int row = threadIdx.x >> 3, seg = threadIdx.x & 7;
    const float* srow = S + row * 1024 + seg * 128;
    _Float16* prow = P + row * 1024 + seg * 128;
    float mx = -1e30f;
    for (int i = 0; i < 128; ++i) mx = fmaxf(mx, srow[i]);
#pragma unroll
    for (int m = 4; m >= 1; m >>= 1) mx = fmaxf(mx, __shfl_xor(mx, m, 8));
    float sum = 0.f;
    for (int i = 0; i < 128; ++i) {
      float p = __expf(srow[i] - mx);
      prow[i] = (_Float16)p;
      sum += p;
    }
#pragma unroll
    for (int m = 4; m >= 1; m >>= 1) sum += __shfl_xor(sum, m, 8);
    if (seg == 0) inv_sum[row] = 1.f / sum;
  }
  __syncthreads();

  // ---- Phase 3: O = P @ V, A-fragments straight from f16 LDS (ds_load_b128)
  {
    const int rt2 = wave >> 2, dct = wave & 3;
    v8f oacc = {};
    for (int ks = 0; ks < W_LEN; ks += 32) {
      v16h a = make_a_frag(P + (rt2 * 16 + am) * 1024 + ks, lane);
      v16h b = make_b_frag(Vb + (size_t)(dct * 16 + am) * W_LEN + ks, lane);
      oacc = wmma_f16(a, b, oacc);
    }
    const int bidx = bh >> 3, h = bh & 7;
    const int dcol = dct * 16 + am;
#pragma unroll
    for (int r = 0; r < 8; ++r) {
      const int rl = rt2 * 16 + r + mhalf;
      const int tok = (w0 + rl) * BSZ + bidx;
      attn_out[(size_t)tok * EMBED + h * HEAD_DIM + dcol] =
          (_Float16)(oacc[r] * inv_sum[rl]);
    }
  }
}

// ---------------------------------------------------------------- out proj + residual
__global__ __launch_bounds__(256) void out_gemm_wmma(
    const _Float16* __restrict__ ao, const _Float16* __restrict__ wout,
    const float* __restrict__ ob, const float* __restrict__ feat,
    float* __restrict__ out) {
  __shared__ alignas(64) _Float16 sA[2 * 64 * 32];
  __shared__ alignas(64) _Float16 sB[2 * 128 * 32];
  const int lane = threadIdx.x & 31, wave = threadIdx.x >> 5;
  v8f acc[2][2] = {};
  gemm_mainloop_64x128(ao, NTOK, wout, EMBED, blockIdx.y * 64, blockIdx.x * 128,
                       sA, sB, lane, wave, acc);
  const int am = lane & 15, mhalf = (lane >> 4) << 3;
  const int m0 = blockIdx.y * 64 + (wave >> 2) * 32;
  const int n0 = blockIdx.x * 128 + (wave & 3) * 32;
#pragma unroll
  for (int i = 0; i < 2; ++i)
#pragma unroll
    for (int j = 0; j < 2; ++j) {
      const int ncol = n0 + j * 16 + am;
      const float bb = ob[ncol];
#pragma unroll
      for (int r = 0; r < 8; ++r) {
        const size_t idx = (size_t)(m0 + i * 16 + r + mhalf) * EMBED + ncol;
        out[idx] = acc[i][j][r] + bb + feat[idx];
      }
    }
}

// ---------------------------------------------------------------------------
extern "C" void kernel_launch(void* const* d_in, const int* in_sizes, int n_in,
                              void* d_out, int out_size, void* d_ws,
                              size_t ws_size, hipStream_t stream) {
  (void)in_sizes; (void)n_in; (void)out_size; (void)ws_size;
  const float* feat      = (const float*)d_in[0];
  const float* in_proj_w = (const float*)d_in[1];
  const float* in_proj_b = (const float*)d_in[2];
  const float* out_w     = (const float*)d_in[3];
  const float* out_b     = (const float*)d_in[4];
  const float* ln_g      = (const float*)d_in[5];
  const float* ln_b      = (const float*)d_in[6];
  float* out = (float*)d_out;

  char* ws = (char*)d_ws;
  size_t off = 0;
  auto take = [&](size_t bytes) {
    char* p = ws + off;
    off += (bytes + 255) & ~size_t(255);
    return p;
  };
  _Float16* xln    = (_Float16*)take((size_t)NTOK * EMBED * 2);
  _Float16* wqkv_h = (_Float16*)take((size_t)QKV_N * EMBED * 2);
  _Float16* wout_h = (_Float16*)take((size_t)EMBED * EMBED * 2);
  _Float16* Qh     = (_Float16*)take((size_t)NBH * W_LEN * HEAD_DIM * 2);
  _Float16* Kh     = (_Float16*)take((size_t)NBH * W_LEN * HEAD_DIM * 2);
  _Float16* Vt     = (_Float16*)take((size_t)NBH * HEAD_DIM * W_LEN * 2);
  _Float16* ao     = (_Float16*)take((size_t)NTOK * EMBED * 2);

  cvt_kernel<<<(QKV_N * EMBED + 255) / 256, 256, 0, stream>>>(in_proj_w, wqkv_h,
                                                              QKV_N * EMBED);
  cvt_kernel<<<(EMBED * EMBED + 255) / 256, 256, 0, stream>>>(out_w, wout_h,
                                                              EMBED * EMBED);
  ln_kernel<<<NTOK / 8, 256, 0, stream>>>(feat, ln_g, ln_b, xln);

  qkv_gemm_wmma<<<dim3(QKV_N / 128, NTOK / 64), 256, 0, stream>>>(
      xln, wqkv_h, in_proj_b, Qh, Kh, Vt);

  const size_t smem = 32 * 1024 * 4 + 32 * 1024 * 2 + 128;  // S + P + inv_sum
  (void)hipFuncSetAttribute((const void*)attn_kernel,
                            hipFuncAttributeMaxDynamicSharedMemorySize,
                            (int)smem);
  attn_kernel<<<dim3(NBH, W_LEN / 32), 256, smem, stream>>>(Qh, Kh, Vt, ao);

  out_gemm_wmma<<<dim3(EMBED / 128, NTOK / 64), 256, 0, stream>>>(
      ao, wout_h, out_b, feat, out);
}
